// Attention_34462817583395
// MI455X (gfx1250) — compile-verified, matrix operand order checked by reading, not run on hardware
//
#include <hip/hip_runtime.h>

// ---------------------------------------------------------------------------
// MI455X (gfx1250) additive-attention forward.
// wave32; bf16 WMMA 16x16x32 (f32 accumulate) for all three GEMMs;
// GLOBAL_LOAD_ASYNC_TO_LDS_B128 for the conversion-free staging paths.
// ---------------------------------------------------------------------------

typedef __attribute__((ext_vector_type(16))) __bf16 v16bf;
typedef __attribute__((ext_vector_type(8)))  float  v8f;

#define NHEAD 8
#define HD    128
#define EMB   1024
#define MB    16
#define SEQ   1024
#define HBD   (NHEAD * MB)   // 128 head-batch rows
#define PITCH 40             // LDS row pitch in halves (32 data + 8 pad); 80B rows => 16B aligned

static __device__ __forceinline__ unsigned short f2bf(float f) {
  unsigned u = __float_as_uint(f);
  u += 0x7FFFu + ((u >> 16) & 1u);     // round-to-nearest-even
  return (unsigned short)(u >> 16);
}

// Per-lane async 16B copy global -> LDS (ASYNCcnt-tracked DMA, no VGPR data).
static __device__ __forceinline__ void async_copy16(void* ldsDst, const void* gsrc) {
  asm volatile("global_load_async_to_lds_b128 %0, %1, off"
               :: "v"((unsigned)(unsigned long long)ldsDst), "v"(gsrc)
               : "memory");
}
static __device__ __forceinline__ void wait_async0() {
  asm volatile("s_wait_asynccnt 0x0" ::: "memory");
}

// Load one 16x32 bf16 fragment (A layout; identical pattern serves B when the
// LDS tile is stored [N][K] row-major).  ISA 7.12.2: lane L -> M=L&15,
// half=L>>4 adds +8 to K; VGPR r holds K pair {2r,2r+1} (r<4) / {16+2(r-4),..}.
static __device__ __forceinline__ v16bf load_frag(const unsigned short* lds, int rowBase) {
  const int lane = threadIdx.x & 31;
  const int half = lane >> 4;
  const int m    = lane & 15;
  const unsigned short* p = lds + (rowBase + m) * PITCH + half * 8;
  union { v16bf v; unsigned int u[8]; } f;
#pragma unroll
  for (int r = 0; r < 8; ++r) {
    const int kb = (r < 4) ? (2 * r) : (16 + 2 * (r - 4));
    f.u[r] = *(const unsigned int*)(p + kb);
  }
  return f.v;
}

// One 128x128 (M x N) tile step over a K-chunk of 32: 8 waves x 8 N-tiles.
static __device__ __forceinline__ void wmma_step(const unsigned short* As,
                                                 const unsigned short* Bs,
                                                 v8f acc[8]) {
  const int wave = threadIdx.x >> 5;
  v16bf a = load_frag(As, wave * 16);
#pragma unroll
  for (int n = 0; n < 8; ++n) {
    v16bf b = load_frag(Bs, n * 16);
    acc[n] = __builtin_amdgcn_wmma_f32_16x16x32_bf16(false, a, false, b,
                                                     (short)0, acc[n], false, false);
  }
}

// Stage a 128-row x 32-col tile (f32 global, row-major) -> bf16 LDS tile (VALU convert).
static __device__ __forceinline__ void stage_f32(unsigned short* dst,
                                                 const float* src, int srcStride) {
  const int tid = threadIdx.x;
  const int c4  = (tid & 7) * 4;
  const int r0  = tid >> 3;
#pragma unroll
  for (int rr = 0; rr < 4; ++rr) {
    const int r = r0 + rr * 32;
    const float4 v = *(const float4*)(src + (size_t)r * srcStride + c4);
    ushort4 wv;
    wv.x = f2bf(v.x); wv.y = f2bf(v.y); wv.z = f2bf(v.z); wv.w = f2bf(v.w);
    *(ushort4*)(dst + r * PITCH + c4) = wv;
  }
}

// Stage a 128-row x 32-col bf16 tile via async DMA: 256 lanes x 2 rows x 16B.
// Caller must wait_async0() + __syncthreads() before consuming the tile.
static __device__ __forceinline__ void stage_bf16_async(unsigned short* dst,
                                                        const unsigned short* src,
                                                        int srcStride) {
  const int tid = threadIdx.x;
  const int c8  = (tid & 3) * 8;       // column chunk in halves (16B)
  const int r0  = tid >> 2;            // 0..63
#pragma unroll
  for (int rr = 0; rr < 2; ++rr) {
    const int r = r0 + rr * 64;
    async_copy16(dst + r * PITCH + c8, src + (size_t)r * srcStride + c8);
  }
}

// exp(tanh(x)) via two v_exp_f32 + one rcp; tanh in (-1,1) => no softmax-max needed.
static __device__ __forceinline__ float score_fn(float x) {
  const float a2 = __expf(2.f * x);
  const float th = 1.f - 2.f / (a2 + 1.f);
  return __expf(th);
}

// --------------------------- K1: weight folding ----------------------------
// vq[h][e] = sum_j Wq[h*H+j][e] * w[H+j]   (kills the entire qx GEMM)
__global__ __launch_bounds__(256) void vq_kernel(const float* __restrict__ Wq,
                                                 const float* __restrict__ w,
                                                 float* __restrict__ vq) {
  const int gid = blockIdx.x * 256 + threadIdx.x;     // 0..8191
  const int h = gid >> 10, e = gid & 1023;
  float acc = 0.f;
#pragma unroll 4
  for (int j = 0; j < HD; ++j)
    acc += Wq[(size_t)(h * HD + j) * EMB + e] * w[HD + j];
  vq[gid] = acc;
}

__global__ void cq_kernel(const float* __restrict__ bq, const float* __restrict__ w,
                          float* __restrict__ cq) {
  const int h = threadIdx.x;
  if (h < NHEAD) {
    float acc = 0.f;
    for (int j = 0; j < HD; ++j) acc += bq[h * HD + j] * w[HD + j];
    cq[h] = acc;
  }
}

// --------------------------- K2: sq GEMV -----------------------------------
// sq[h*MB+b][t] = q[b][t][:] . vq[h][:] + cq[h]; one wave per row, all heads.
__global__ __launch_bounds__(256) void sq_kernel(const float* __restrict__ q,
                                                 const float* __restrict__ vq,
                                                 const float* __restrict__ cq,
                                                 float* __restrict__ sq) {
  const int b    = blockIdx.y;
  const int wave = threadIdx.x >> 5;
  const int lane = threadIdx.x & 31;
  const int t    = blockIdx.x * 8 + wave;
  const float* qr = q + ((size_t)b * SEQ + t) * EMB;
  float acc[NHEAD] = {};
  for (int e = lane; e < EMB; e += 32) {
    const float qv = qr[e];
#pragma unroll
    for (int h = 0; h < NHEAD; ++h) acc[h] += qv * vq[h * EMB + e];
  }
#pragma unroll
  for (int h = 0; h < NHEAD; ++h) {
#pragma unroll
    for (int m = 16; m > 0; m >>= 1) acc[h] += __shfl_xor(acc[h], m, 32);
  }
  if (lane < NHEAD)
    sq[(size_t)(lane * MB + b) * SEQ + t] = acc[lane] + cq[lane];
}

// --------------------------- K3: kx GEMM + sk ------------------------------
// kx[hb][t][j] = k[b][t][:] . Wk[h*H+j][:] + bk[h*H+j]
// writes kxT bf16 [hb][j][s] (so attention's B staging is coalesced) and
// sk[hb][t] = kx . w[:H] via in-register shuffle reduction.
__global__ __launch_bounds__(256) void kx_kernel(const float* __restrict__ k,
                                                 const float* __restrict__ Wk,
                                                 const float* __restrict__ bk,
                                                 const float* __restrict__ w,
                                                 unsigned short* __restrict__ kxT,
                                                 float* __restrict__ sk) {
  __shared__ unsigned short As[128 * PITCH];
  __shared__ unsigned short Bs[128 * PITCH];
  const int hb = blockIdx.y, h = hb >> 4, b = hb & 15;
  const int t0 = blockIdx.x * 128;
  const float* A  = k  + ((size_t)b * SEQ + t0) * EMB;
  const float* Bw = Wk + (size_t)h * HD * EMB;
  const v8f zero = {0.f, 0.f, 0.f, 0.f, 0.f, 0.f, 0.f, 0.f};
  v8f acc[8];
#pragma unroll
  for (int n = 0; n < 8; ++n) acc[n] = zero;

  for (int kk = 0; kk < EMB; kk += 32) {
    stage_f32(As, A + kk, EMB);
    stage_f32(Bs, Bw + kk, EMB);
    __syncthreads();
    wmma_step(As, Bs, acc);
    __syncthreads();
  }

  const int wave = threadIdx.x >> 5, lane = threadIdx.x & 31;
  const int half = lane >> 4, n0 = lane & 15;
  float skp[8] = {};
#pragma unroll
  for (int n = 0; n < 8; ++n) {
    const int j = n * 16 + n0;
    const float bias = bk[h * HD + j];
    const float wj   = w[j];
#pragma unroll
    for (int r = 0; r < 8; ++r) {
      const float v = acc[n][r] + bias;
      skp[r] += v * wj;
      const int t = t0 + wave * 16 + r + 8 * half;
      kxT[((size_t)hb * HD + j) * SEQ + t] = f2bf(v);
    }
  }
#pragma unroll
  for (int r = 0; r < 8; ++r) {
#pragma unroll
    for (int m = 8; m > 0; m >>= 1) skp[r] += __shfl_xor(skp[r], m, 32);  // within 16-lane half
    if (n0 == 0)
      sk[(size_t)hb * SEQ + t0 + wave * 16 + r + 8 * half] = skp[r];
  }
}

// --------------------------- K4: attention ---------------------------------
// score[hb][t][s] = exp(tanh(sq+sk)) / den  (written f32 to d_out, bf16 to LDS)
// out_mid = score @ kx  -> out_cat bf16 [b][t][h*H+j]
// B tile arrives via async DMA overlapped with the VALU score computation.
__global__ __launch_bounds__(256) void attn_kernel(const unsigned short* __restrict__ kxT,
                                                   const float* __restrict__ sk,
                                                   const float* __restrict__ sq,
                                                   float* __restrict__ score_out,
                                                   unsigned short* __restrict__ outcat) {
  __shared__ unsigned short As[128 * PITCH];
  __shared__ unsigned short Bs[128 * PITCH];
  __shared__ float skl[SEQ];
  __shared__ float sql[128];
  __shared__ float rdn[128];
  const int hb = blockIdx.y, h = hb >> 4, b = hb & 15;
  const int t0 = blockIdx.x * 128;
  const int tid = threadIdx.x, wave = tid >> 5, lane = tid & 31;

  for (int i = tid; i < SEQ; i += 256) skl[i] = sk[(size_t)hb * SEQ + i];
  if (tid < 128) sql[tid] = sq[(size_t)hb * SEQ + t0 + tid];
  __syncthreads();

  // Pass 1: softmax denominators (one wave per 16 rows, lanes stride over s).
  for (int m = 0; m < 16; ++m) {
    const int row = wave * 16 + m;
    const float sqv = sql[row];
    float s = 0.f;
    for (int j = lane; j < SEQ; j += 32) s += score_fn(sqv + skl[j]);
#pragma unroll
    for (int mm = 16; mm > 0; mm >>= 1) s += __shfl_xor(s, mm, 32);
    if (lane == 0) rdn[row] = 1.f / s;
  }
  __syncthreads();

  const v8f zero = {0.f, 0.f, 0.f, 0.f, 0.f, 0.f, 0.f, 0.f};
  v8f acc[8];
#pragma unroll
  for (int n = 0; n < 8; ++n) acc[n] = zero;

  const int c4 = (tid & 7) * 4, r0 = tid >> 3;
  for (int kk = 0; kk < SEQ; kk += 32) {
    // Issue async DMA for B tile (kx[s][j] stored as [j][s]) ...
    stage_bf16_async(Bs, kxT + (size_t)hb * HD * SEQ + kk, SEQ);
    // ... and overlap it with the VALU-heavy score tile (A).
#pragma unroll
    for (int rr = 0; rr < 4; ++rr) {
      const int r = r0 + rr * 32;
      const float sqv = sql[r];
      const float rd  = rdn[r];
      const float v0 = score_fn(sqv + skl[kk + c4 + 0]) * rd;
      const float v1 = score_fn(sqv + skl[kk + c4 + 1]) * rd;
      const float v2 = score_fn(sqv + skl[kk + c4 + 2]) * rd;
      const float v3 = score_fn(sqv + skl[kk + c4 + 3]) * rd;
      ushort4 wv;
      wv.x = f2bf(v0); wv.y = f2bf(v1); wv.z = f2bf(v2); wv.w = f2bf(v3);
      *(ushort4*)(As + r * PITCH + c4) = wv;
      *(float4*)(score_out + ((size_t)hb * SEQ + t0 + r) * SEQ + kk + c4) =
          make_float4(v0, v1, v2, v3);
    }
    wait_async0();
    __syncthreads();
    wmma_step(As, Bs, acc);
    __syncthreads();   // protects Bs from next iteration's async writes
  }

  const int half = lane >> 4, n0 = lane & 15;
#pragma unroll
  for (int n = 0; n < 8; ++n) {
#pragma unroll
    for (int r = 0; r < 8; ++r) {
      const int t = t0 + wave * 16 + r + 8 * half;
      const int j = n * 16 + n0;
      outcat[((size_t)b * SEQ + t) * EMB + h * HD + j] = f2bf(acc[n][r]);
    }
  }
}

// --------------------------- K5: output projection -------------------------
// out[b][t][e] = out_cat[b][t][:] . Wp[e][:] + bp[e]
// A tile arrives via async DMA overlapped with the Wp f32->bf16 VALU staging.
__global__ __launch_bounds__(256) void proj_kernel(const unsigned short* __restrict__ outcat,
                                                   const float* __restrict__ Wp,
                                                   const float* __restrict__ bp,
                                                   float* __restrict__ out) {
  __shared__ unsigned short As[128 * PITCH];
  __shared__ unsigned short Bs[128 * PITCH];
  const int row0 = blockIdx.y * 128;   // rows of (b,t): 16384 total
  const int e0   = blockIdx.x * 128;
  const v8f zero = {0.f, 0.f, 0.f, 0.f, 0.f, 0.f, 0.f, 0.f};
  v8f acc[8];
#pragma unroll
  for (int n = 0; n < 8; ++n) acc[n] = zero;

  for (int kk = 0; kk < EMB; kk += 32) {
    stage_bf16_async(As, outcat + (size_t)row0 * EMB + kk, EMB);
    stage_f32(Bs, Wp + (size_t)e0 * EMB + kk, EMB);
    wait_async0();
    __syncthreads();
    wmma_step(As, Bs, acc);
    __syncthreads();
  }

  const int wave = threadIdx.x >> 5, lane = threadIdx.x & 31;
  const int half = lane >> 4, n0 = lane & 15;
#pragma unroll
  for (int n = 0; n < 8; ++n) {
    const int e = e0 + n * 16 + n0;
    const float bias = bp[e];
#pragma unroll
    for (int r = 0; r < 8; ++r) {
      const int row = row0 + wave * 16 + r + 8 * half;
      out[(size_t)row * EMB + e] = acc[n][r] + bias;
    }
  }
}

// ---------------------------------------------------------------------------
extern "C" void kernel_launch(void* const* d_in, const int* in_sizes, int n_in,
                              void* d_out, int out_size, void* d_ws, size_t ws_size,
                              hipStream_t stream) {
  (void)in_sizes; (void)n_in; (void)out_size; (void)ws_size;
  const float* k  = (const float*)d_in[0];
  const float* q  = (const float*)d_in[1];
  const float* Wk = (const float*)d_in[2];
  const float* bk = (const float*)d_in[3];
  const float* Wq = (const float*)d_in[4];
  const float* bq = (const float*)d_in[5];
  const float* w  = (const float*)d_in[6];
  const float* Wp = (const float*)d_in[7];
  const float* bp = (const float*)d_in[8];

  float* out   = (float*)d_out;
  float* score = out + (size_t)MB * SEQ * EMB;      // second tuple element

  char* ws = (char*)d_ws;
  size_t off = 0;
  unsigned short* kxT    = (unsigned short*)(ws + off); off += (size_t)HBD * HD * SEQ * 2; // 32 MB
  unsigned short* outcat = (unsigned short*)(ws + off); off += (size_t)MB * SEQ * EMB * 2; // 32 MB
  float* skw = (float*)(ws + off); off += (size_t)HBD * SEQ * 4;                            // 512 KB
  float* sqw = (float*)(ws + off); off += (size_t)HBD * SEQ * 4;                            // 512 KB
  float* vqw = (float*)(ws + off); off += (size_t)NHEAD * EMB * 4;                          // 32 KB
  float* cqw = (float*)(ws + off); off += 64;

  vq_kernel  <<<dim3(32),                          dim3(256), 0, stream>>>(Wq, w, vqw);
  cq_kernel  <<<dim3(1),                           dim3(32),  0, stream>>>(bq, w, cqw);
  sq_kernel  <<<dim3(SEQ / 8, MB),                 dim3(256), 0, stream>>>(q, vqw, cqw, sqw);
  kx_kernel  <<<dim3(SEQ / 128, HBD),              dim3(256), 0, stream>>>(k, Wk, bk, w, kxT, skw);
  attn_kernel<<<dim3(SEQ / 128, HBD),              dim3(256), 0, stream>>>(kxT, skw, sqw, score, outcat);
  proj_kernel<<<dim3(EMB / 128, (MB * SEQ) / 128), dim3(256), 0, stream>>>(outcat, Wp, bp, out);
}
